// StarTransformer_11287174054023
// MI455X (gfx1250) — compile-verified, matrix operand order checked by reading, not run on hardware
//
#include <hip/hip_runtime.h>
#include <hip/hip_bf16.h>
#include <stdint.h>

typedef __attribute__((ext_vector_type(16))) _Float16 v16h;
typedef __attribute__((ext_vector_type(8)))  _Float16 v8h;
typedef __attribute__((ext_vector_type(8)))  float    v8f;

#define B_    16
#define L_    2048
#define BL_   32768      // B_*L_
#define LP1_  2049
#define RY_   32784      // B_*(L_+1)
#define H_    300
#define HP_   320        // padded hidden
#define NH_   6
#define HD_   50
#define NL_   3
#define KT_   10         // K tiles (320/32)
#define NT_   20         // N tiles (320/16)
#define NG_   5          // N groups of 4 tiles (64 cols) per wave
#define SCALE_ 0.14142135623730951f   // 1/sqrt(50)
#define PACK_ELEMS 102400             // KT_*HP_*32 f16 per weight matrix

// ---------------------------------------------------------------------------
// Weight pre-pack: W[out=n][in=k] (300x300 f32) -> WMMA B-fragment order f16.
// Element layout: idx = ((kt*320 + n)*2 + hi)*16 + i,  k = kt*32 + (i&7) + hi*8 + (i>>3)*16
// ---------------------------------------------------------------------------
__global__ void pack_w(const float* __restrict__ W, _Float16* __restrict__ Bp) {
  int idx = blockIdx.x * blockDim.x + threadIdx.x;
  if (idx >= PACK_ELEMS) return;
  int i  = idx & 15;
  int hi = (idx >> 4) & 1;
  int n  = (idx >> 5) % HP_;
  int kt = (idx >> 5) / HP_;
  int k  = kt * 32 + (i & 7) + hi * 8 + (i >> 3) * 16;
  float v = (n < H_ && k < H_) ? W[n * H_ + k] : 0.0f;
  Bp[idx] = (_Float16)v;
}

// ---------------------------------------------------------------------------
// WMMA GEMM: C[M x 320] = A[M x 320](f16) * Wpack + bias, optional leaky-relu.
// One wave per 16x64 tile (4 WMMA accumulators, A fragment reused 4x).
// 8 waves per block. M must be a multiple of 16.
// ---------------------------------------------------------------------------
__global__ void gemm_wmma(const _Float16* __restrict__ A,
                          const _Float16* __restrict__ Bp,
                          const float* __restrict__ bias,
                          float* __restrict__ C,
                          int Mtiles, int relu) {
  int wave = (blockIdx.x * blockDim.x + threadIdx.x) >> 5;
  int lane = threadIdx.x & 31;
  int total = Mtiles * NG_;
  if (wave >= total) return;                  // uniform per-wave, EXEC stays full
  int mTile = wave / NG_;
  int g     = wave % NG_;                     // group of 4 consecutive N tiles
  int lo = lane & 15;
  int hi = lane >> 4;

  const _Float16* arow = A  + (size_t)(mTile * 16 + lo) * HP_ + hi * 8;
  // base of the 4 packed B columns for this lane (subtiles are +512 f16 apart)
  const _Float16* bcol = Bp + (size_t)((g * 4) * 16 + lo) * 32 + hi * 16;

  v8f acc0 = {}, acc1 = {}, acc2 = {}, acc3 = {};
#pragma unroll
  for (int kt = 0; kt < KT_; ++kt) {
    v8h a0 = *(const v8h*)(arow + kt * 32);
    v8h a1 = *(const v8h*)(arow + kt * 32 + 16);
    v16h av = __builtin_shufflevector(a0, a1, 0,1,2,3,4,5,6,7,8,9,10,11,12,13,14,15);

    const _Float16* bk = bcol + (size_t)kt * HP_ * 32;
    if (kt + 1 < KT_)  // hint next K-step's packed-B line into cache
      __builtin_prefetch(bcol + (size_t)(kt + 1) * HP_ * 32, 0, 0);

    v8h b0, b1; v16h bv;
    b0 = *(const v8h*)(bk);          b1 = *(const v8h*)(bk + 8);
    bv = __builtin_shufflevector(b0, b1, 0,1,2,3,4,5,6,7,8,9,10,11,12,13,14,15);
    acc0 = __builtin_amdgcn_wmma_f32_16x16x32_f16(false, av, false, bv, (short)0, acc0, false, false);

    b0 = *(const v8h*)(bk + 512);    b1 = *(const v8h*)(bk + 520);
    bv = __builtin_shufflevector(b0, b1, 0,1,2,3,4,5,6,7,8,9,10,11,12,13,14,15);
    acc1 = __builtin_amdgcn_wmma_f32_16x16x32_f16(false, av, false, bv, (short)0, acc1, false, false);

    b0 = *(const v8h*)(bk + 1024);   b1 = *(const v8h*)(bk + 1032);
    bv = __builtin_shufflevector(b0, b1, 0,1,2,3,4,5,6,7,8,9,10,11,12,13,14,15);
    acc2 = __builtin_amdgcn_wmma_f32_16x16x32_f16(false, av, false, bv, (short)0, acc2, false, false);

    b0 = *(const v8h*)(bk + 1536);   b1 = *(const v8h*)(bk + 1544);
    bv = __builtin_shufflevector(b0, b1, 0,1,2,3,4,5,6,7,8,9,10,11,12,13,14,15);
    acc3 = __builtin_amdgcn_wmma_f32_16x16x32_f16(false, av, false, bv, (short)0, acc3, false, false);
  }

  int mbase = mTile * 16 + hi * 8;
  v8f accs[4] = {acc0, acc1, acc2, acc3};
#pragma unroll
  for (int t = 0; t < 4; ++t) {
    int n = (g * 4 + t) * 16 + lo;
    float bn = (n < H_) ? bias[n] : 0.0f;
#pragma unroll
    for (int r = 0; r < 8; ++r) {
      float v = accs[t][r] + bn;
      if (relu) v = (v < 0.0f) ? 0.01f * v : v;
      C[(size_t)(mbase + r) * HP_ + n] = v;
    }
  }
}

// ---------------------------------------------------------------------------
// Embedding build: nodes32 (f32 padded) and embed16 (f16 padded).
// ---------------------------------------------------------------------------
__global__ void embed_build(const int* __restrict__ x,
                            const float* __restrict__ tab,
                            const float* __restrict__ pos,
                            _Float16* __restrict__ e16,
                            float* __restrict__ nodes32) {
  int idx = blockIdx.x * blockDim.x + threadIdx.x;
  if (idx >= BL_ * HP_) return;
  int r = idx / HP_, c = idx % HP_;
  float v = 0.0f;
  if (c < H_) {
    int l = r % L_;
    v = tab[(size_t)x[r] * H_ + c] + pos[(size_t)l * H_ + c];
  }
  e16[idx] = (_Float16)v;
  nodes32[idx] = v;
}

// relay = mean over L of embed
__global__ void relay_init(const int* __restrict__ x,
                           const float* __restrict__ tab,
                           const float* __restrict__ pos,
                           float* __restrict__ relay) {
  int idx = blockIdx.x * blockDim.x + threadIdx.x;
  if (idx >= B_ * HP_) return;
  int b = idx / HP_, c = idx % HP_;
  float s = 0.0f;
  if (c < H_) {
    for (int l = 0; l < L_; ++l)
      s += tab[(size_t)x[b * L_ + l] * H_ + c] + pos[(size_t)l * H_ + c];
  }
  relay[idx] = s * (1.0f / (float)L_);
}

// ---------------------------------------------------------------------------
// LayerNorm over first 300 cols -> f16 padded output. One block (128) per row.
// ---------------------------------------------------------------------------
__global__ void ln_to_f16(const float* __restrict__ X, _Float16* __restrict__ Y,
                          const float* __restrict__ g, const float* __restrict__ b) {
  __shared__ float s1[128], s2[128];
  int row = blockIdx.x;
  const float* xr = X + (size_t)row * HP_;
  float s = 0.0f, ss = 0.0f;
  for (int c = threadIdx.x; c < H_; c += 128) { float v = xr[c]; s += v; ss += v * v; }
  s1[threadIdx.x] = s; s2[threadIdx.x] = ss;
  __syncthreads();
  for (int o = 64; o > 0; o >>= 1) {
    if (threadIdx.x < o) { s1[threadIdx.x] += s1[threadIdx.x + o]; s2[threadIdx.x] += s2[threadIdx.x + o]; }
    __syncthreads();
  }
  float mean = s1[0] * (1.0f / H_);
  float var  = s2[0] * (1.0f / H_) - mean * mean;
  float rstd = rsqrtf(var + 1e-6f);
  for (int c = threadIdx.x; c < HP_; c += 128) {
    float v = (c < H_) ? (xr[c] - mean) * rstd * g[c] + b[c] : 0.0f;
    Y[(size_t)row * HP_ + c] = (_Float16)v;
  }
}

// plain f32 -> f16
__global__ void conv16(const float* __restrict__ X, _Float16* __restrict__ Y, int n) {
  int i = blockIdx.x * blockDim.x + threadIdx.x;
  if (i < n) Y[i] = (_Float16)X[i];
}

// y = concat([relay, nodes], axis=1) -> f16 rows [B*(L+1)][320]
__global__ void build_y16(const float* __restrict__ relay,
                          const float* __restrict__ nodes,
                          _Float16* __restrict__ y) {
  int idx = blockIdx.x * blockDim.x + threadIdx.x;
  if (idx >= RY_ * HP_) return;
  int row = idx / HP_, c = idx % HP_;
  int b = row / LP1_, j = row % LP1_;
  float v = (j == 0) ? relay[b * HP_ + c]
                     : nodes[(size_t)(b * L_ + j - 1) * HP_ + c];
  y[idx] = (_Float16)v;
}

// ---------------------------------------------------------------------------
// Ring attention: 5 candidates {k[l-1],k[l],k[l+1],ak_e[l],ak_r[b]} per head.
// Block = 192 threads = 6 waves = 6 heads, one block per (b,l).
// att may alias q: each head only touches its own 50-column slice.
// ---------------------------------------------------------------------------
__global__ void ring_scores(const float* __restrict__ q, const float* __restrict__ k,
                            const float* __restrict__ v,
                            const float* __restrict__ ak, const float* __restrict__ av,
                            const float* __restrict__ akr, const float* __restrict__ avr,
                            float* __restrict__ att) {
  int row = blockIdx.x;            // b*L + l
  int b = row >> 11;               // /2048
  int l = row & (L_ - 1);
  int h = threadIdx.x >> 5;
  int lane = threadIdx.x & 31;
  int base = h * HD_;
  const float* qp = q + (size_t)row * HP_ + base;

  auto dot = [&](const float* kp) -> float {
    float s = 0.0f;
    for (int c = lane; c < HD_; c += 32) s += qp[c] * kp[c];
    for (int o = 16; o > 0; o >>= 1) s += __shfl_xor(s, o, 32);
    return s;
  };

  float s0 = (l > 0)      ? dot(k + (size_t)(row - 1) * HP_ + base) : 0.0f;  // zero-pad: dot with 0
  float s1 = dot(k + (size_t)row * HP_ + base);
  float s2 = (l < L_ - 1) ? dot(k + (size_t)(row + 1) * HP_ + base) : 0.0f;
  float s3 = dot(ak + (size_t)row * HP_ + base);
  float s4 = dot(akr + (size_t)b * HP_ + base);
  s0 *= SCALE_; s1 *= SCALE_; s2 *= SCALE_; s3 *= SCALE_; s4 *= SCALE_;

  float mx = fmaxf(fmaxf(fmaxf(s0, s1), fmaxf(s2, s3)), s4);
  float e0 = __expf(s0 - mx), e1 = __expf(s1 - mx), e2 = __expf(s2 - mx);
  float e3 = __expf(s3 - mx), e4 = __expf(s4 - mx);
  float inv = 1.0f / (e0 + e1 + e2 + e3 + e4);
  e0 *= inv; e1 *= inv; e2 *= inv; e3 *= inv; e4 *= inv;

  for (int c = lane; c < HD_; c += 32) {
    float acc = e1 * v[(size_t)row * HP_ + base + c]
              + e3 * av[(size_t)row * HP_ + base + c]
              + e4 * avr[(size_t)b * HP_ + base + c];
    if (l > 0)      acc += e0 * v[(size_t)(row - 1) * HP_ + base + c];
    if (l < L_ - 1) acc += e2 * v[(size_t)(row + 1) * HP_ + base + c];
    att[(size_t)row * HP_ + base + c] = acc;
  }
  if (h == 0 && lane < HP_ - H_) att[(size_t)row * HP_ + H_ + lane] = 0.0f;
}

// ---------------------------------------------------------------------------
// Star attention: relay attends over 2049 keys. One block (256) per (b, head).
// ---------------------------------------------------------------------------
__global__ void star_attn(const float* __restrict__ sq, const float* __restrict__ sk,
                          const float* __restrict__ sv, const unsigned char* __restrict__ mask,
                          float* __restrict__ satt) {
  __shared__ float sqs[HD_];
  __shared__ float sc[LP1_];
  __shared__ float red[256];
  int b = blockIdx.x / NH_;
  int h = blockIdx.x % NH_;
  int base = h * HD_;
  int tid = threadIdx.x;

  if (tid < HD_) sqs[tid] = sq[b * HP_ + base + tid];
  __syncthreads();

  float lm = -1e30f;
  for (int j = tid; j < LP1_; j += 256) {
    float s;
    bool padj = (j > 0) && (mask[b * L_ + j - 1] == 0);
    if (padj) {
      s = -1e30f;
    } else {
      s = 0.0f;
      const float* kp = sk + (size_t)(b * LP1_ + j) * HP_ + base;
      for (int c = 0; c < HD_; ++c) s += sqs[c] * kp[c];
      s *= SCALE_;
    }
    sc[j] = s;
    lm = fmaxf(lm, s);
  }
  red[tid] = lm; __syncthreads();
  for (int o = 128; o > 0; o >>= 1) {
    if (tid < o) red[tid] = fmaxf(red[tid], red[tid + o]);
    __syncthreads();
  }
  float mx = red[0];
  __syncthreads();

  float ls = 0.0f;
  for (int j = tid; j < LP1_; j += 256) { float e = __expf(sc[j] - mx); sc[j] = e; ls += e; }
  __syncthreads();
  red[tid] = ls; __syncthreads();
  for (int o = 128; o > 0; o >>= 1) {
    if (tid < o) red[tid] += red[tid + o];
    __syncthreads();
  }
  float inv = 1.0f / red[0];
  __syncthreads();

  if (tid < HD_) {
    float acc = 0.0f;
    for (int j = 0; j < LP1_; ++j)
      acc += sc[j] * sv[(size_t)(b * LP1_ + j) * HP_ + base + tid];
    satt[b * HP_ + base + tid] = acc * inv;
  }
  if (h == 0 && tid >= 64 && tid < 64 + (HP_ - H_)) satt[b * HP_ + H_ + (tid - 64)] = 0.0f;
}

// nodes = where(pad, 0, nodes)
__global__ void mask_nodes(const unsigned char* __restrict__ mask, float* __restrict__ nodes) {
  int idx = blockIdx.x * blockDim.x + threadIdx.x;
  if (idx >= BL_ * HP_) return;
  int r = idx / HP_;
  if (mask[r] == 0) nodes[idx] = 0.0f;
}

// Emit: out = nodes[B,L,300] flat ++ relay[B,300] flat
__global__ void emit_out(const float* __restrict__ nodes, const float* __restrict__ relay,
                         float* __restrict__ out) {
  int idx = blockIdx.x * blockDim.x + threadIdx.x;
  const int n_nodes = BL_ * H_;
  const int total = n_nodes + B_ * H_;
  if (idx >= total) return;
  if (idx < n_nodes) {
    int r = idx / H_, c = idx % H_;
    out[idx] = nodes[(size_t)r * HP_ + c];
  } else {
    int t = idx - n_nodes;
    int b = t / H_, c = t % H_;
    out[idx] = relay[b * HP_ + c];
  }
}

// ---------------------------------------------------------------------------
// Host launcher
// ---------------------------------------------------------------------------
extern "C" void kernel_launch(void* const* d_in, const int* in_sizes, int n_in,
                              void* d_out, int out_size, void* d_ws, size_t ws_size,
                              hipStream_t stream) {
  const int*            x    = (const int*)d_in[0];
  const unsigned char*  mask = (const unsigned char*)d_in[1];
  const float*          tab  = (const float*)d_in[2];
  const float*          pos  = (const float*)d_in[3];
  const float*          ng   = (const float*)d_in[4];
  const float*          nb   = (const float*)d_in[5];
  // d_in[6 + 2*w] = weight w, d_in[7 + 2*w] = bias w
  // w: 0=ring_Wq 1=ring_Wk 2=ring_Wv 3=ring_Wo 4=star_Wq 5=star_Wk 6=star_Wv 7=star_Wo

  char* ws = (char*)d_ws;
  const size_t BIG32 = (size_t)RY_ * HP_ * 4;   // 41,963,520
  const size_t BIG16 = (size_t)RY_ * HP_ * 2;
  const size_t SM32  = (size_t)B_ * HP_ * 4;    // 20,480
  const size_t SM16  = (size_t)B_ * HP_ * 2;

  size_t off = 0;
  _Float16* wpack   = (_Float16*)(ws + off); off += (size_t)NL_ * 8 * PACK_ELEMS * 2;
  _Float16* embed16 = (_Float16*)(ws + off); off += BIG16;
  float*    nodes32 = (float*)(ws + off);    off += BIG32;
  _Float16* a16     = (_Float16*)(ws + off); off += BIG16;   // nx16 / att16 / y16 staging
  float*    q32     = (float*)(ws + off);    off += BIG32;   // also att32 (aliased)
  float*    k32     = (float*)(ws + off);    off += BIG32;   // also sk32
  float*    v32     = (float*)(ws + off);    off += BIG32;   // also sv32
  float*    ak32    = (float*)(ws + off);    off += BIG32;
  float*    av32    = (float*)(ws + off);    off += BIG32;
  float*    relay32 = (float*)(ws + off);    off += SM32;
  float*    akr32   = (float*)(ws + off);    off += SM32;
  float*    avr32   = (float*)(ws + off);    off += SM32;
  float*    sq32    = (float*)(ws + off);    off += SM32;
  float*    satt32  = (float*)(ws + off);    off += SM32;
  _Float16* rbuf16  = (_Float16*)(ws + off); off += SM16;
  float*    att32   = q32;                                    // alias (safe: per-head slices)

  auto gemm = [&](const _Float16* A, int matId, const float* bias, float* C,
                  int M, int relu) {
    int mt = M / 16;
    int waves = mt * NG_;
    int blocks = (waves + 7) / 8;
    gemm_wmma<<<blocks, 256, 0, stream>>>(A, wpack + (size_t)matId * PACK_ELEMS,
                                          bias, C, mt, relu);
  };

  // ---- pack all 24 weight matrices to f16 WMMA-B fragment order ----
  for (int i = 0; i < NL_; ++i)
    for (int w = 0; w < 8; ++w) {
      const float* W = (const float*)d_in[6 + 2 * w] + (size_t)i * H_ * H_;
      pack_w<<<(PACK_ELEMS + 255) / 256, 256, 0, stream>>>(W, wpack + (size_t)(i * 8 + w) * PACK_ELEMS);
    }

  // ---- embedding, relay init ----
  embed_build<<<(BL_ * HP_ + 255) / 256, 256, 0, stream>>>(x, tab, pos, embed16, nodes32);
  relay_init<<<(B_ * HP_ + 255) / 256, 256, 0, stream>>>(x, tab, pos, relay32);

  for (int i = 0; i < NL_; ++i) {
    const float* g  = ng + i * H_;
    const float* bb = nb + i * H_;
    int mWq = i * 8 + 0, mWk = i * 8 + 1, mWv = i * 8 + 2, mWo = i * 8 + 3;
    int mSq = i * 8 + 4, mSk = i * 8 + 5, mSv = i * 8 + 6, mSo = i * 8 + 7;
    const float* r_bq = (const float*)d_in[7]  + i * H_;
    const float* r_bk = (const float*)d_in[9]  + i * H_;
    const float* r_bv = (const float*)d_in[11] + i * H_;
    const float* r_bo = (const float*)d_in[13] + i * H_;
    const float* s_bq = (const float*)d_in[15] + i * H_;
    const float* s_bk = (const float*)d_in[17] + i * H_;
    const float* s_bv = (const float*)d_in[19] + i * H_;
    const float* s_bo = (const float*)d_in[21] + i * H_;

    // ring attention
    ln_to_f16<<<BL_, 128, 0, stream>>>(nodes32, a16, g, bb);            // nx
    gemm(a16, mWq, r_bq, q32, BL_, 0);
    gemm(a16, mWk, r_bk, k32, BL_, 0);
    gemm(a16, mWv, r_bv, v32, BL_, 0);
    gemm(embed16, mWk, r_bk, ak32, BL_, 0);                             // ak_e
    gemm(embed16, mWv, r_bv, av32, BL_, 0);                             // av_e
    conv16<<<(B_ * HP_ + 255) / 256, 256, 0, stream>>>(relay32, rbuf16, B_ * HP_);
    gemm(rbuf16, mWk, r_bk, akr32, B_, 0);                              // ak_r
    gemm(rbuf16, mWv, r_bv, avr32, B_, 0);                              // av_r
    ring_scores<<<BL_, 192, 0, stream>>>(q32, k32, v32, ak32, av32, akr32, avr32, att32);
    conv16<<<(BL_ * HP_ + 255) / 256, 256, 0, stream>>>(att32, a16, BL_ * HP_);
    gemm(a16, mWo, r_bo, nodes32, BL_, 1);                              // leaky-relu

    // star attention
    ln_to_f16<<<B_, 128, 0, stream>>>(relay32, rbuf16, g, bb);          // rx
    gemm(rbuf16, mSq, s_bq, sq32, B_, 0);
    build_y16<<<(RY_ * HP_ + 255) / 256, 256, 0, stream>>>(relay32, nodes32, a16);
    gemm(a16, mSk, s_bk, k32, RY_, 0);                                  // sk
    gemm(a16, mSv, s_bv, v32, RY_, 0);                                  // sv
    star_attn<<<B_ * NH_, 256, 0, stream>>>(sq32, k32, v32, mask, satt32);
    conv16<<<(B_ * HP_ + 255) / 256, 256, 0, stream>>>(satt32, rbuf16, B_ * HP_);
    gemm(rbuf16, mSo, s_bo, relay32, B_, 1);                            // leaky-relu

    mask_nodes<<<(BL_ * HP_ + 255) / 256, 256, 0, stream>>>(mask, nodes32);
  }

  int total_out = BL_ * H_ + B_ * H_;
  emit_out<<<(total_out + 255) / 256, 256, 0, stream>>>(nodes32, relay32, (float*)d_out);
}
